// SelfAttention2d_31791347925640
// MI455X (gfx1250) — matrix operand order repaired and checked by reading, not compiled, not run on hardware
//
#include <hip/hip_runtime.h>

// ---------------------------------------------------------------------------
// Types for CDNA5 WMMA (wave32): bf16 inputs, f32 accumulate, 16x16x32 shape.
// ---------------------------------------------------------------------------
typedef __attribute__((ext_vector_type(16))) __bf16 v16bf;
typedef __attribute__((ext_vector_type(8)))  __bf16 v8bf;
typedef __attribute__((ext_vector_type(8)))  float  v8f;

#define LANE (threadIdx.x & 31)
#define WAVE (threadIdx.x >> 5)

static constexpr int CH    = 512;
static constexpr int NTOK  = 1024;   // 32*32 spatial
static constexpr int HEADS = 8;
static constexpr int HD    = 64;     // CH / HEADS
static constexpr int B     = 8;

// WMMA wrapper: D = A(16x32 bf16) * B(32x16 bf16) + C(16x16 f32)
__device__ __forceinline__ v8f wmma_bf16(v16bf a, v16bf b, v8f c) {
    return __builtin_amdgcn_wmma_f32_16x16x32_bf16(
        /*neg_a=*/false, a, /*neg_b=*/false, b,
        /*c_mod=*/(short)0, c, /*reuse_a=*/false, /*reuse_b=*/false);
}

// A-fragment (16 rows x 32 K), source row-major [row][k], ld in elements.
// Layout: lane<16 -> row=lane, K 0..15; lane>=16 -> same rows, K 16..31
// (matches load_bfrag).  => two contiguous 16B loads.
__device__ __forceinline__ v16bf load_afrag(const __bf16* base, int ld, int lane) {
    int row  = lane & 15;
    int koff = (lane >> 4) * 16;
    const __bf16* p = base + (size_t)row * ld + koff;
    v8bf lo = *(const v8bf*)(p);
    v8bf hi = *(const v8bf*)(p + 8);
    return __builtin_shufflevector(lo, hi, 0,1,2,3,4,5,6,7,8,9,10,11,12,13,14,15);
}

// B-fragment (32 K x 16 cols), source stored K-contiguous per column:
// [col][k] row-major.  lane<16 -> col=lane, K 0..15; lane>=16 -> K 16..31.
// => one contiguous 32B load per lane.
__device__ __forceinline__ v16bf load_bfrag(const __bf16* base, int ld, int lane) {
    int col  = lane & 15;
    int koff = (lane >> 4) * 16;
    return *(const v16bf*)(base + (size_t)col * ld + koff);
}

// ---------------------------------------------------------------------------
// 1) f32 -> bf16 weight conversion
// ---------------------------------------------------------------------------
__global__ void cvt_bf16(const float* __restrict__ src, __bf16* __restrict__ dst, int n) {
    int i = blockIdx.x * 256 + threadIdx.x;
    if (i < n) dst[i] = (__bf16)src[i];
}

// ---------------------------------------------------------------------------
// 2) GroupNorm: one block per (b, group). Normalize 64ch x 1024 spatial,
//    write bf16 TRANSPOSED activations xnT[b][n][c] for K-contiguous GEMM A.
// ---------------------------------------------------------------------------
__global__ void gn_kernel(const float* __restrict__ x,
                          const float* __restrict__ gw,
                          const float* __restrict__ gb,
                          __bf16* __restrict__ xnT) {
    const int b = blockIdx.x >> 3;
    const int g = blockIdx.x & 7;
    const float* xp = x + ((size_t)b * CH + g * 64) * NTOK;

    float s = 0.f, s2 = 0.f;
    for (int i = threadIdx.x; i < 64 * NTOK; i += 256) {
        float v = xp[i];
        s += v; s2 += v * v;
    }
    __shared__ float rs[256], rs2[256];
    rs[threadIdx.x] = s; rs2[threadIdx.x] = s2;
    __syncthreads();
    for (int o = 128; o > 0; o >>= 1) {
        if (threadIdx.x < o) { rs[threadIdx.x] += rs[threadIdx.x + o];
                               rs2[threadIdx.x] += rs2[threadIdx.x + o]; }
        __syncthreads();
    }
    const float inv  = 1.f / (64.f * NTOK);
    const float mean = rs[0] * inv;
    const float var  = rs2[0] * inv - mean * mean;
    const float rstd = rsqrtf(var + 1e-5f);

    // write transposed, c fastest so stores are contiguous (x re-reads hit L2)
    for (int i = threadIdx.x; i < 64 * NTOK; i += 256) {
        int cc = i & 63, n = i >> 6;
        int c  = g * 64 + cc;
        float v = (xp[(size_t)cc * NTOK + n] - mean) * rstd * gw[c] + gb[c];
        xnT[((size_t)b * NTOK + n) * CH + c] = (__bf16)v;
    }
}

// ---------------------------------------------------------------------------
// 3) QKV GEMM: qkv[b][n][o] = bias[o] + sum_c xnT[b][n][c] * Wq[o][c]
//    q slice (o<512) pre-scaled by 1/sqrt(hd).
//    Block = 128(n) x 64(o), 8 waves each 32x32 (2x2 WMMA tiles).
// ---------------------------------------------------------------------------
__global__ void qkv_gemm(const __bf16* __restrict__ xnT,
                         const __bf16* __restrict__ Wq,   // [3C][C] row-major
                         const float*  __restrict__ bias, // [3C]
                         __bf16* __restrict__ qkv) {      // [B][N][3C]
    const int lane = LANE, w = WAVE;
    const int b  = blockIdx.z;
    const int n0 = blockIdx.y * 128 + (w >> 1) * 32;
    const int o0 = blockIdx.x * 64  + (w & 1)  * 32;

    const __bf16* A  = xnT + ((size_t)b * NTOK + n0) * CH;
    const __bf16* Bm = Wq  + (size_t)o0 * CH;

    v8f acc[2][2] = {};
    for (int k = 0; k < CH; k += 32) {
        v16bf a0 = load_afrag(A + k,             CH, lane);
        v16bf a1 = load_afrag(A + 16 * CH + k,   CH, lane);
        v16bf b0 = load_bfrag(Bm + k,            CH, lane);
        v16bf b1 = load_bfrag(Bm + 16 * CH + k,  CH, lane);
        acc[0][0] = wmma_bf16(a0, b0, acc[0][0]);
        acc[0][1] = wmma_bf16(a0, b1, acc[0][1]);
        acc[1][0] = wmma_bf16(a1, b0, acc[1][0]);
        acc[1][1] = wmma_bf16(a1, b1, acc[1][1]);
    }
    // store: for fixed r, lanes 0-15 cover 16 consecutive bf16 (32B segment)
    const int half = lane >> 4, col = lane & 15;
    for (int rt = 0; rt < 2; ++rt)
        for (int ct = 0; ct < 2; ++ct) {
            int o = o0 + ct * 16 + col;
            float bo = bias[o];
            float scale = (o < CH) ? 0.125f : 1.0f;  // 1/sqrt(64) on q
            for (int r = 0; r < 8; ++r) {
                int n = n0 + rt * 16 + r + 8 * half;
                float v = (acc[rt][ct][r] + bo) * scale;
                qkv[((size_t)b * NTOK + n) * (3 * CH) + o] = (__bf16)v;
            }
        }
}

// ---------------------------------------------------------------------------
// 4) Flash attention: block = (b, h, 128 q-rows); wave owns 16 q-rows with
//    resident q A-fragments; streams key/value tiles of 32 tokens.
// ---------------------------------------------------------------------------
__global__ void attn_kernel(const __bf16* __restrict__ qkv,  // [B][N][3C]
                            __bf16* __restrict__ ao) {       // [B][N][C]
    const int lane = LANE, w = WAVE;
    const int b = blockIdx.z, h = blockIdx.y;
    const int q0 = blockIdx.x * 128 + w * 16;
    const int LD = 3 * CH;

    // resident q fragments (K = 64 -> two 32-wide k-steps)
    const __bf16* qb = qkv + ((size_t)b * NTOK + q0) * LD + h * HD;
    v16bf aq0 = load_afrag(qb,      LD, lane);
    v16bf aq1 = load_afrag(qb + 32, LD, lane);

    __shared__ __align__(64) __bf16 VT[HD * 32];        // V tile transposed [c][m]
    __shared__ __align__(64) __bf16 P[8][16 * 32];      // per-wave prob tiles

    float mrun[8], lrun[8];
    v8f acc[4] = {};
    for (int r = 0; r < 8; ++r) { mrun[r] = -1e30f; lrun[r] = 0.f; }

    const int half = lane >> 4, col = lane & 15;

    // per-thread V-staging coordinates: 8 threads cover one token's 128B slice
    const int vm = threadIdx.x >> 3;        // token within tile: 0..31
    const int vc = (threadIdx.x & 7) * 8;   // channel offset: 0..56

    for (int m0 = 0; m0 < NTOK; m0 += 32) {
        __syncthreads();                 // previous VT fully consumed
        {   // stage V tile transposed: one coalesced b128 load per thread,
            // scatter into VT[c][m] with ds_store_b16
            const __bf16* vsrc =
                qkv + ((size_t)b * NTOK + m0 + vm) * LD + 2 * CH + h * HD + vc;
            v8bf vv = *(const v8bf*)vsrc;
            for (int j = 0; j < 8; ++j)
                VT[(vc + j) * 32 + vm] = vv[j];
            // prefetch next tile's V slice (global_prefetch_b8 path)
            if (m0 + 32 < NTOK)
                __builtin_prefetch(vsrc + 32 * LD, 0, 0);
        }
        __syncthreads();                 // VT ready

        // scores tile 16x32 = two 16x16 WMMA outputs, K=64 (2 steps)
        v8f S[2] = {};
        for (int st = 0; st < 2; ++st) {
            const __bf16* kb = qkv + ((size_t)b * NTOK + m0 + st * 16) * LD + CH + h * HD;
            v16bf bk0 = load_bfrag(kb,      LD, lane);
            v16bf bk1 = load_bfrag(kb + 32, LD, lane);
            S[st] = wmma_bf16(aq0, bk0, S[st]);
            S[st] = wmma_bf16(aq1, bk1, S[st]);
            if (m0 + 32 < NTOK)          // prefetch next K tile rows
                __builtin_prefetch(kb + 32 * LD, 0, 0);
        }

        // online softmax: row r occupies one 16-lane half at register r
        __bf16* Pw = &P[w][0];
        for (int r = 0; r < 8; ++r) {
            float mx = fmaxf(S[0][r], S[1][r]);
            for (int msk = 1; msk < 16; msk <<= 1)
                mx = fmaxf(mx, __shfl_xor(mx, msk, 32));
            float mnew = fmaxf(mrun[r], mx);
            float cf   = __expf(mrun[r] - mnew);
            float p0   = __expf(S[0][r] - mnew);
            float p1   = __expf(S[1][r] - mnew);
            float rsum = p0 + p1;
            for (int msk = 1; msk < 16; msk <<= 1)
                rsum += __shfl_xor(rsum, msk, 32);
            lrun[r] = lrun[r] * cf + rsum;
            mrun[r] = mnew;
            for (int ct = 0; ct < 4; ++ct) acc[ct][r] *= cf;
            int rowi = r + 8 * half;
            Pw[rowi * 32 + col]      = (__bf16)p0;
            Pw[rowi * 32 + 16 + col] = (__bf16)p1;
        }

        // P(16x32) @ V(32x64): one k-step, 4 column tiles
        v16bf ap = load_afrag(Pw, 32, lane);
        for (int ct = 0; ct < 4; ++ct) {
            v16bf bv = load_bfrag(VT + ct * 16 * 32, 32, lane);
            acc[ct] = wmma_bf16(ap, bv, acc[ct]);
        }
    }

    // epilogue: normalize and write token-major bf16 (32B segments per half-wave)
    for (int ct = 0; ct < 4; ++ct) {
        int c = h * HD + ct * 16 + col;
        for (int r = 0; r < 8; ++r) {
            int n = q0 + r + 8 * half;
            ao[((size_t)b * NTOK + n) * CH + c] = (__bf16)(acc[ct][r] / lrun[r]);
        }
    }
}

// ---------------------------------------------------------------------------
// 5) Proj GEMM + residual: y[b][o][n] = x[b][o][n] + pb[o] + sum_c ao[b][n][c]*Wp[o][c]
//    Epilogue transposes each wave's 32x32 tile through LDS so the channel-
//    major f32 output is written as contiguous 128B rows per lane.
// ---------------------------------------------------------------------------
__global__ void proj_gemm(const __bf16* __restrict__ ao,   // [B][N][C]
                          const __bf16* __restrict__ Wp,   // [C][C]
                          const float*  __restrict__ pb,   // [C]
                          const float*  __restrict__ x,    // [B][C][N]
                          float* __restrict__ out) {       // [B][C][N]
    const int lane = LANE, w = WAVE;
    const int b  = blockIdx.z;
    const int n0 = blockIdx.y * 128 + (w >> 1) * 32;
    const int o0 = blockIdx.x * 64  + (w & 1)  * 32;

    const __bf16* A  = ao + ((size_t)b * NTOK + n0) * CH;
    const __bf16* Bm = Wp + (size_t)o0 * CH;

    v8f acc[2][2] = {};
    for (int k = 0; k < CH; k += 32) {
        v16bf a0 = load_afrag(A + k,            CH, lane);
        v16bf a1 = load_afrag(A + 16 * CH + k,  CH, lane);
        v16bf b0 = load_bfrag(Bm + k,           CH, lane);
        v16bf b1 = load_bfrag(Bm + 16 * CH + k, CH, lane);
        acc[0][0] = wmma_bf16(a0, b0, acc[0][0]);
        acc[0][1] = wmma_bf16(a0, b1, acc[0][1]);
        acc[1][0] = wmma_bf16(a1, b0, acc[1][0]);
        acc[1][1] = wmma_bf16(a1, b1, acc[1][1]);
    }

    // transpose wave tile through LDS: stage[o_local][n_local], stride 36
    // (keeps float4 16B alignment and spreads banks)
    __shared__ __align__(16) float stage[8][32 * 36];
    float* st = stage[w];
    const int half = lane >> 4, col = lane & 15;
    for (int rt = 0; rt < 2; ++rt)
        for (int ct = 0; ct < 2; ++ct)
            for (int r = 0; r < 8; ++r) {
                int ol = ct * 16 + col;            // o within tile
                int nl = rt * 16 + r + 8 * half;   // n within tile
                st[ol * 36 + nl] = acc[rt][ct][r];
            }
    // per-wave region, same-wave DS ops are in-order: no barrier needed.
    // each lane owns one o-row: contiguous 128B residual read + store
    {
        int o = o0 + lane;
        float bo = pb[o];
        size_t rowbase = ((size_t)b * CH + o) * NTOK + n0;
        const float* xr = x + rowbase;
        float* orow     = out + rowbase;
        for (int j = 0; j < 32; j += 4) {
            float4 xv = *(const float4*)(xr + j);
            float4 rv;
            rv.x = xv.x + st[lane * 36 + j + 0] + bo;
            rv.y = xv.y + st[lane * 36 + j + 1] + bo;
            rv.z = xv.z + st[lane * 36 + j + 2] + bo;
            rv.w = xv.w + st[lane * 36 + j + 3] + bo;
            *(float4*)(orow + j) = rv;
        }
    }
}

// ---------------------------------------------------------------------------
// Host launcher
// ---------------------------------------------------------------------------
extern "C" void kernel_launch(void* const* d_in, const int* in_sizes, int n_in,
                              void* d_out, int out_size, void* d_ws, size_t ws_size,
                              hipStream_t stream) {
    const float* x      = (const float*)d_in[0];
    const float* gn_w   = (const float*)d_in[1];
    const float* gn_b   = (const float*)d_in[2];
    const float* qkv_w  = (const float*)d_in[3];
    const float* qkv_b  = (const float*)d_in[4];
    const float* proj_w = (const float*)d_in[5];
    const float* proj_b = (const float*)d_in[6];
    float* out = (float*)d_out;

    // workspace carve-out (all sizes multiples of 256B)
    char* p = (char*)d_ws;
    __bf16* wq_bf  = (__bf16*)p; p += (size_t)3 * CH * CH * 2;        // 1.5 MB
    __bf16* wp_bf  = (__bf16*)p; p += (size_t)CH * CH * 2;            // 0.5 MB
    __bf16* xnT    = (__bf16*)p; p += (size_t)B * NTOK * CH * 2;      // 8 MB
    __bf16* qkvbuf = (__bf16*)p; p += (size_t)B * NTOK * 3 * CH * 2;  // 24 MB
    __bf16* aobuf  = (__bf16*)p;                                      // 8 MB

    cvt_bf16<<<(3 * CH * CH + 255) / 256, 256, 0, stream>>>(qkv_w, wq_bf, 3 * CH * CH);
    cvt_bf16<<<(CH * CH + 255) / 256, 256, 0, stream>>>(proj_w, wp_bf, CH * CH);

    gn_kernel<<<B * 8, 256, 0, stream>>>(x, gn_w, gn_b, xnT);

    // qkv: grid (o-tiles=24, n-tiles=8, b=8)
    qkv_gemm<<<dim3(3 * CH / 64, NTOK / 128, B), 256, 0, stream>>>(xnT, wq_bf, qkv_b, qkvbuf);

    // attention: grid (q-tiles=8, heads=8, b=8)
    attn_kernel<<<dim3(NTOK / 128, HEADS, B), 256, 0, stream>>>(qkvbuf, aobuf);

    // proj + residual: grid (o-tiles=8, n-tiles=8, b=8)
    proj_gemm<<<dim3(CH / 64, NTOK / 128, B), 256, 0, stream>>>(aobuf, wp_bf, proj_b, x, out);
}
